// MultiHeadAttention_46265387713219
// MI455X (gfx1250) — matrix operand order repaired and checked by reading, not compiled, not run on hardware
//
#include <hip/hip_runtime.h>
#include <hip/hip_bf16.h>
#include <math.h>

typedef __bf16 bf16_t;
typedef __attribute__((ext_vector_type(16))) __bf16 v16bf;
typedef __attribute__((ext_vector_type(8)))  __bf16 v8bf;
typedef __attribute__((ext_vector_type(4)))  __bf16 v4bf;
typedef __attribute__((ext_vector_type(8)))  float  v8f;

#define WMMA_BF16(a, b, c) \
  __builtin_amdgcn_wmma_f32_16x16x32_bf16(false, (a), false, (b), (short)0, (c), false, false)

// A-fragment loader for 16x32 bf16 tiles (row = lane&15, K-half = lane>>4).
// p must point at &M[row][k0 + 8*(lane>>4)]; elements e<8 come from +0,
// e>=8 from +16 (per ISA 16-bit A-matrix layout).
static __device__ inline v16bf load_a_frag(const bf16_t* p) {
  v8bf lo = *(const v8bf*)p;
  v8bf hi = *(const v8bf*)(p + 16);
  v16bf r;
#pragma unroll
  for (int i = 0; i < 8; ++i) { r[i] = lo[i]; r[8 + i] = hi[i]; }
  return r;
}

// Generic-to-LDS byte address: for flat shared pointers the LDS offset lives
// in addr[31:0] (aperture in the upper half), so truncation yields the value
// the async-DMA VDST VGPR expects.
static __device__ inline uint32_t lds_addr32(const void* p) {
  return (uint32_t)(uintptr_t)p;
}

// Async global->LDS 16B copy (tracked by ASYNCcnt). Inline asm so it works on
// both toolchains regardless of builtin availability.
static __device__ inline void async_g2l_b128(uint32_t lds_byte_addr, const void* gptr) {
  asm volatile("global_load_async_to_lds_b128 %0, %1, off"
               :: "v"(lds_byte_addr), "v"(gptr)
               : "memory");
}

static __device__ inline void wait_asynccnt0() {
  asm volatile("s_wait_asynccnt 0x0" ::: "memory");
}

// ---------------------------------------------------------------------------
// Weight transpose + f32->bf16:  Wt[n*D + k] = (bf16) W[k*D + n]
// ---------------------------------------------------------------------------
__global__ __launch_bounds__(256)
void transpose_w_kernel(const float* __restrict__ W, bf16_t* __restrict__ Wt, int D) {
  __shared__ bf16_t tile[32][33];
  int n0 = blockIdx.x * 32, k0 = blockIdx.y * 32;
  int tx = threadIdx.x, ty = threadIdx.y;
#pragma unroll
  for (int r = ty; r < 32; r += 8)
    tile[r][tx] = (bf16_t)W[(size_t)(k0 + r) * D + n0 + tx];
  __syncthreads();
#pragma unroll
  for (int r = ty; r < 32; r += 8)
    Wt[(size_t)(n0 + r) * D + k0 + tx] = tile[tx][r];
}

// ---------------------------------------------------------------------------
// f32 -> bf16 bulk convert (n divisible by 4)
// ---------------------------------------------------------------------------
__global__ __launch_bounds__(256)
void cvt_bf16_kernel(const float* __restrict__ in, bf16_t* __restrict__ out, int n) {
  int i = (blockIdx.x * blockDim.x + threadIdx.x) * 4;
  if (i < n) {
    float4 v = *(const float4*)(in + i);
    v4bf o;
    o[0] = (bf16_t)v.x; o[1] = (bf16_t)v.y; o[2] = (bf16_t)v.z; o[3] = (bf16_t)v.w;
    *(v4bf*)(out + i) = o;
  }
}

// ---------------------------------------------------------------------------
// WMMA GEMM: Y = X(8192xD bf16) * W(DxD, stored transposed Wt[n][k]) + bias
// mode 0/1: write bf16 head-split [b,h,s,d] (mode 0 pre-scales by out_scale)
// mode 2  : write bf16 d-major    [b,h,d,s]  (for V)
// mode 3  : write f32 row-major   [row, col] (final output)
// WG = 256 thr (8 waves), WG tile 128x128, wave tile 32x64, K-step 32.
// ---------------------------------------------------------------------------
__global__ __launch_bounds__(256)
void gemm_wmma_kernel(const bf16_t* __restrict__ X, const bf16_t* __restrict__ Wt,
                      const float* __restrict__ bias, void* __restrict__ Yout,
                      int mode, float out_scale) {
  const int D = 1024, S = 2048;
  int lane = threadIdx.x & 31;
  int wave = threadIdx.x >> 5;
  int h = lane >> 4, ln = lane & 15;
  int m0 = blockIdx.y * 128 + (wave & 3) * 32;
  int n0 = blockIdx.x * 128 + (wave >> 2) * 64;

  v8f zero = {0.f, 0.f, 0.f, 0.f, 0.f, 0.f, 0.f, 0.f};
  v8f acc[2][4];
#pragma unroll
  for (int i = 0; i < 2; ++i)
#pragma unroll
    for (int j = 0; j < 4; ++j) acc[i][j] = zero;

  const bf16_t* Arow0 = X + (size_t)(m0 + ln) * D + 8 * h;
  const bf16_t* Arow1 = X + (size_t)(m0 + 16 + ln) * D + 8 * h;
  const bf16_t* Brow[4];
#pragma unroll
  for (int j = 0; j < 4; ++j)
    Brow[j] = Wt + (size_t)(n0 + j * 16 + ln) * D + 8 * h;

  for (int kk = 0; kk < D; kk += 32) {
    v16bf a0 = load_a_frag(Arow0 + kk);
    v16bf a1 = load_a_frag(Arow1 + kk);
#pragma unroll
    for (int j = 0; j < 4; ++j) {
      v16bf b = load_a_frag(Brow[j] + kk);
      acc[0][j] = WMMA_BF16(a0, b, acc[0][j]);
      acc[1][j] = WMMA_BF16(a1, b, acc[1][j]);
    }
  }

#pragma unroll
  for (int am = 0; am < 2; ++am) {
#pragma unroll
    for (int bn = 0; bn < 4; ++bn) {
      int c = n0 + bn * 16 + ln;        // output column
      float bv = bias[c];
      int rbase = m0 + am * 16 + 8 * h; // first of 8 consecutive rows
      if (mode == 2) {                  // Vt[b][h][d][s]
        int hh = c >> 6, d = c & 63;
        int b_ = rbase >> 11, sq = rbase & 2047;
        v8bf pk;
#pragma unroll
        for (int i = 0; i < 8; ++i)
          pk[i] = (bf16_t)((acc[am][bn][i] + bv) * out_scale);
        bf16_t* dst = (bf16_t*)Yout + ((size_t)((b_ * 16 + hh) * 64 + d)) * S + sq;
        *(v8bf*)dst = pk;
      } else if (mode == 3) {           // f32 row-major
        float* O = (float*)Yout;
#pragma unroll
        for (int i = 0; i < 8; ++i)
          O[(size_t)(rbase + i) * D + c] = acc[am][bn][i] + bv;
      } else {                          // Qh/Kh [b][h][s][d]
        bf16_t* O = (bf16_t*)Yout;
        int hh = c >> 6, d = c & 63;
#pragma unroll
        for (int i = 0; i < 8; ++i) {
          int r = rbase + i;
          int b_ = r >> 11, sq = r & 2047;
          O[(((size_t)(b_ * 16 + hh) * S) + sq) * 64 + d] =
              (bf16_t)((acc[am][bn][i] + bv) * out_scale);
        }
      }
    }
  }
}

// ---------------------------------------------------------------------------
// Transposed flash attention with async-LDS staging of K / V^T blocks.
// Each wave owns 16 queries (column dim of the transposed score tile -> all
// softmax state is per-lane scalar).
//   S^T = K_blk (32x64) * Q^T (64x16)      [2 key tiles x 2 K-steps = 4 WMMA]
//   O^T = V^T  (64x32)  * P^T (32x16)      [4 d tiles               = 4 WMMA]
// K/V blocks are shared by all 8 waves; they are DMA'd into LDS with
// GLOBAL_LOAD_ASYNC_TO_LDS_B128 (double buffered, one barrier per block),
// pre-swizzled into per-lane A-fragment order so every LDS read is one
// contiguous 32B lane-major access (conflict-free: each bank hit exactly
// twice per wave).
// grid = (B*H, S/128), block = 256 (8 waves).
// ---------------------------------------------------------------------------
__global__ __launch_bounds__(256)
void attention_wmma_kernel(const bf16_t* __restrict__ Qh, const bf16_t* __restrict__ Kh,
                           const bf16_t* __restrict__ Vt, bf16_t* __restrict__ ctx) {
  const int S = 2048, DEP = 64, D = 1024;
  // Fragment-order staging buffers: [buf][frag*32 + lane][16 bf16]
  __shared__ __align__(32) bf16_t Klds[2][4 * 32 * 16];  // 4 frags (t,s)
  __shared__ __align__(32) bf16_t Vlds[2][4 * 32 * 16];  // 4 frags (t)

  int tid = threadIdx.x;
  int lane = tid & 31;
  int wave = tid >> 5;
  int h = lane >> 4, ln = lane & 15;
  int bh = blockIdx.x;
  int q0 = blockIdx.y * 128 + wave * 16;

  const bf16_t* Q = Qh + (size_t)bh * S * DEP;
  const bf16_t* K = Kh + (size_t)bh * S * DEP;
  const bf16_t* V = Vt + (size_t)bh * DEP * S;

  // Per-thread staging assignment (256 threads <-> 256 16B chunks per matrix).
  // K chunk: t = tid>>7, s = (tid>>6)&1, lane_ = (tid>>1)&31, c = tid&1
  int kc_c = tid & 1, kc_lane = (tid >> 1) & 31, kc_s = (tid >> 6) & 1, kc_t = tid >> 7;
  int kc_h = kc_lane >> 4, kc_ln = kc_lane & 15;
  size_t kg_off = (size_t)(kc_t * 16 + kc_ln) * DEP + kc_s * 32 + 16 * kc_h + 8 * kc_c;
  uint32_t klds_elem = (uint32_t)((((kc_t * 2 + kc_s) * 32 + kc_lane) * 2 + kc_c) * 8);
  // V chunk: t = tid>>6, lane_ = (tid>>1)&31, c = tid&1
  int vc_c = tid & 1, vc_lane = (tid >> 1) & 31, vc_t = tid >> 6;
  int vc_h = vc_lane >> 4, vc_ln = vc_lane & 15;
  size_t vg_off = (size_t)(vc_t * 16 + vc_ln) * S + 16 * vc_h + 8 * vc_c;
  uint32_t vlds_elem = (uint32_t)(((vc_t * 32 + vc_lane) * 2 + vc_c) * 8);

  // Q^T B-fragments: element e -> depth (e + 16h + 32*step), column = query ln
  v16bf qb0 = *(const v16bf*)(Q + (size_t)(q0 + ln) * DEP + 16 * h);
  v16bf qb1 = *(const v16bf*)(Q + (size_t)(q0 + ln) * DEP + 32 + 16 * h);

  v8f zero = {0.f, 0.f, 0.f, 0.f, 0.f, 0.f, 0.f, 0.f};
  v8f acc[4];
#pragma unroll
  for (int t = 0; t < 4; ++t) acc[t] = zero;
  float mrun = -INFINITY, lrun = 0.f;

  // Prologue: stage key block 0 into buffer 0.
  async_g2l_b128(lds_addr32(&Klds[0][klds_elem]), K + (size_t)0 * DEP + kg_off);
  async_g2l_b128(lds_addr32(&Vlds[0][vlds_elem]), V + 0 + vg_off);
  wait_asynccnt0();
  __syncthreads();

  for (int kb = 0, it = 0; kb < S; kb += 32, ++it) {
    int cur = it & 1, nxt = cur ^ 1;
    if (kb + 32 < S) {  // prefetch next key block into the other buffer
      async_g2l_b128(lds_addr32(&Klds[nxt][klds_elem]),
                     K + (size_t)(kb + 32) * DEP + kg_off);
      async_g2l_b128(lds_addr32(&Vlds[nxt][vlds_elem]),
                     V + (kb + 32) + vg_off);
    }

    // S^T = K_blk * Q^T  -- fragments are contiguous 32B per lane in LDS.
    v8f st0 = zero, st1 = zero;
    v16bf ka;
    ka = *(const v16bf*)(&Klds[cur][(0 * 32 + lane) * 16]);  // t=0, s=0
    st0 = WMMA_BF16(ka, qb0, st0);
    ka = *(const v16bf*)(&Klds[cur][(1 * 32 + lane) * 16]);  // t=0, s=1
    st0 = WMMA_BF16(ka, qb1, st0);
    ka = *(const v16bf*)(&Klds[cur][(2 * 32 + lane) * 16]);  // t=1, s=0
    st1 = WMMA_BF16(ka, qb0, st1);
    ka = *(const v16bf*)(&Klds[cur][(3 * 32 + lane) * 16]);  // t=1, s=1
    st1 = WMMA_BF16(ka, qb1, st1);

    // Online softmax (query = ln fixed per lane; keys split across lane^16)
    float bm = -INFINITY;
#pragma unroll
    for (int i = 0; i < 8; ++i) bm = fmaxf(bm, fmaxf(st0[i], st1[i]));
    bm = fmaxf(bm, __shfl_xor(bm, 16));
    float mnew = fmaxf(mrun, bm);
    float alpha = __expf(mrun - mnew);
    float p0[8], p1[8], rs = 0.f;
#pragma unroll
    for (int i = 0; i < 8; ++i) {
      p0[i] = __expf(st0[i] - mnew);
      p1[i] = __expf(st1[i] - mnew);
      rs += p0[i] + p1[i];
    }
    rs += __shfl_xor(rs, 16);
    lrun = lrun * alpha + rs;
    mrun = mnew;
#pragma unroll
    for (int t = 0; t < 4; ++t)
#pragma unroll
      for (int i = 0; i < 8; ++i) acc[t][i] *= alpha;

    // P^T B-fragment: b[e] needs key = e + 16h; D-layout holds key = i + 8h.
    // Half-wave shuffle moves the missing 8-key group from lane^16.
    v16bf pb;
#pragma unroll
    for (int i = 0; i < 8; ++i) {
      float s0 = __shfl_xor(p0[i], 16);
      float s1 = __shfl_xor(p1[i], 16);
      pb[i]     = (bf16_t)(h ? s1 : p0[i]);
      pb[8 + i] = (bf16_t)(h ? p1[i] : s0);
    }
#pragma unroll
    for (int t = 0; t < 4; ++t) {
      v16bf va = *(const v16bf*)(&Vlds[cur][(t * 32 + lane) * 16]);
      acc[t] = WMMA_BF16(va, pb, acc[t]);
    }

    wait_asynccnt0();   // our prefetch into `nxt` has landed
    __syncthreads();    // everyone done reading `cur` / writing `nxt`
  }

  float inv = 1.f / lrun;
  int b_ = bh >> 4, hh = bh & 15;
  int q = q0 + ln;
#pragma unroll
  for (int t = 0; t < 4; ++t) {
    v8bf pk;
#pragma unroll
    for (int i = 0; i < 8; ++i) pk[i] = (bf16_t)(acc[t][i] * inv);
    // O^T frag: row d = t*16 + 8h + i (8 consecutive), col = query
    bf16_t* dst = ctx + ((size_t)(b_ * S + q)) * D + hh * 64 + t * 16 + 8 * h;
    *(v8bf*)dst = pk;
  }
}

// ---------------------------------------------------------------------------
extern "C" void kernel_launch(void* const* d_in, const int* in_sizes, int n_in,
                              void* d_out, int out_size, void* d_ws, size_t ws_size,
                              hipStream_t stream) {
  const int B = 4, S = 2048, D = 1024, H = 16;
  const int M = B * S;                     // 8192 rows
  const size_t WBYTES = (size_t)D * D * 2; // bf16 weight
  const size_t XBYTES = (size_t)M * D * 2; // bf16 activations

  const float* q  = (const float*)d_in[0];
  const float* k  = (const float*)d_in[1];
  const float* v  = (const float*)d_in[2];
  const float* Wq = (const float*)d_in[3];
  const float* bq = (const float*)d_in[4];
  const float* Wk = (const float*)d_in[5];
  const float* bk = (const float*)d_in[6];
  const float* Wv = (const float*)d_in[7];
  const float* bv = (const float*)d_in[8];
  const float* Wo = (const float*)d_in[9];
  const float* bo = (const float*)d_in[10];

  char* ws = (char*)d_ws;
  size_t off = 0;
  auto take = [&](size_t bytes) { size_t o = off; off += (bytes + 255) & ~(size_t)255; return o; };
  bf16_t* wtq = (bf16_t*)(ws + take(WBYTES));
  bf16_t* wtk = (bf16_t*)(ws + take(WBYTES));
  bf16_t* wtv = (bf16_t*)(ws + take(WBYTES));
  bf16_t* wto = (bf16_t*)(ws + take(WBYTES));
  bf16_t* xq  = (bf16_t*)(ws + take(XBYTES));
  bf16_t* xk  = (bf16_t*)(ws + take(XBYTES));
  bf16_t* xv  = (bf16_t*)(ws + take(XBYTES));
  bf16_t* qh  = (bf16_t*)(ws + take(XBYTES));
  bf16_t* kh  = (bf16_t*)(ws + take(XBYTES));
  bf16_t* vt  = (bf16_t*)(ws + take(XBYTES));
  bf16_t* ctx = (bf16_t*)(ws + take(XBYTES));

  dim3 tb(32, 8);
  dim3 tg(D / 32, D / 32);
  transpose_w_kernel<<<tg, tb, 0, stream>>>(Wq, wtq, D);
  transpose_w_kernel<<<tg, tb, 0, stream>>>(Wk, wtk, D);
  transpose_w_kernel<<<tg, tb, 0, stream>>>(Wv, wtv, D);
  transpose_w_kernel<<<tg, tb, 0, stream>>>(Wo, wto, D);

  int n = M * D;
  int cblocks = n / 4 / 256;
  cvt_bf16_kernel<<<cblocks, 256, 0, stream>>>(q, xq, n);
  cvt_bf16_kernel<<<cblocks, 256, 0, stream>>>(k, xk, n);
  cvt_bf16_kernel<<<cblocks, 256, 0, stream>>>(v, xv, n);

  dim3 ggrid(D / 128, M / 128);
  gemm_wmma_kernel<<<ggrid, 256, 0, stream>>>(xq, wtq, bq, qh, 0, 0.125f);  // 1/sqrt(64)
  gemm_wmma_kernel<<<ggrid, 256, 0, stream>>>(xk, wtk, bk, kh, 1, 1.0f);
  gemm_wmma_kernel<<<ggrid, 256, 0, stream>>>(xv, wtv, bv, vt, 2, 1.0f);

  dim3 agrid(B * H, S / 128);
  attention_wmma_kernel<<<agrid, 256, 0, stream>>>(qh, kh, vt, ctx);

  gemm_wmma_kernel<<<ggrid, 256, 0, stream>>>(ctx, wto, bo, d_out, 3, 1.0f);
}
